// HomoGNN_5729486373618
// MI455X (gfx1250) — compile-verified, matrix operand order checked by reading, not run on hardware
//
#include <hip/hip_runtime.h>
#include <math.h>

// ---------------- problem constants (match reference) ----------------
#define NN    50000      // nodes
#define DD    128        // input features
#define HC    256        // heads * per-head width
#define CC    128        // per-head width
#define EE    800000     // edges (before self loops)
#define EP    (EE + NN)  // edges incl. self loops
#define NEG_SLOPE 0.2f

typedef __attribute__((ext_vector_type(16))) __bf16 v16bf;
typedef __attribute__((ext_vector_type(8)))  __bf16 v8bf;
typedef __attribute__((ext_vector_type(8)))  float  v8f;

// ---------------- helpers ----------------
__device__ __forceinline__ void edge_sd(int e, const int* __restrict__ ei,
                                        int& s, int& d) {
    if (e < EE) { s = ei[e]; d = ei[EE + e]; }
    else        { s = e - EE; d = e - EE; }          // self loop
}

__device__ __forceinline__ unsigned enc_f32(float f) {
    unsigned u = __float_as_uint(f);
    return (u & 0x80000000u) ? ~u : (u | 0x80000000u);
}
__device__ __forceinline__ float dec_f32(unsigned u) {
    u = (u & 0x80000000u) ? (u & 0x7FFFFFFFu) : ~u;
    return __uint_as_float(u);
}

// ---------------- conversions ----------------
__global__ void k_f32_to_bf16(const float* __restrict__ in,
                              __bf16* __restrict__ out, int n) {
    int i = blockIdx.x * blockDim.x + threadIdx.x;
    if (i < n) out[i] = (__bf16)in[i];
}

// W: [K][Nn] row-major  ->  Wt: [Nn][K] (bf16) so each output column is K-contiguous
__global__ void k_transpose_bf16(const float* __restrict__ W,
                                 __bf16* __restrict__ Wt, int K, int Nn) {
    int i = blockIdx.x * blockDim.x + threadIdx.x;
    if (i < K * Nn) {
        int n = i / K, k = i - n * K;
        Wt[i] = (__bf16)W[(size_t)k * Nn + n];
    }
}

// ---------------- WMMA GEMM: C[M,Nn] = A[M,K] * Bt[Nn,K]^T  (bf16 in, f32 out) ----
// one wave per 16x16 tile; K stepped by 32 with v_wmma_f32_16x16x32_bf16
__global__ __launch_bounds__(256)
void k_gemm_bf16_wmma(const __bf16* __restrict__ A,
                      const __bf16* __restrict__ Bt,
                      float* __restrict__ C,
                      int M, int Nn, int K) {
    const int wavesPerBlock = blockDim.x >> 5;
    int wave = blockIdx.x * wavesPerBlock + (threadIdx.x >> 5);
    int lane = threadIdx.x & 31;
    int tilesN = Nn >> 4;
    int tM = (wave / tilesN) << 4;
    int tN = (wave % tilesN) << 4;
    if (tM >= M) return;

    int half = lane >> 4;       // lanes 16..31 hold the upper K-halves
    int lr   = lane & 15;

    v8f acc = {};
    const __bf16* arow = A  + (size_t)(tM + lr) * K;
    const __bf16* brow = Bt + (size_t)(tN + lr) * K;

    for (int kk = 0; kk < K; kk += 32) {
        // 16-bit A layout: lanes 0-15 -> K[0..7] in V0..3, K[16..23] in V4..7
        //                  lanes 16-31 -> K[8..15], K[24..31]
        v8bf alo = *(const v8bf*)(arow + kk + 8 * half);
        v8bf ahi = *(const v8bf*)(arow + kk + 16 + 8 * half);
        v8bf blo = *(const v8bf*)(brow + kk + 8 * half);
        v8bf bhi = *(const v8bf*)(brow + kk + 16 + 8 * half);
        v16bf a, b;
#pragma unroll
        for (int i = 0; i < 8; ++i) {
            a[i] = alo[i]; a[i + 8] = ahi[i];
            b[i] = blo[i]; b[i + 8] = bhi[i];
        }
        acc = __builtin_amdgcn_wmma_f32_16x16x32_bf16(
                  false, a, false, b, (short)0, acc, false, false);
    }

    // f32 C/D layout: VGPR r -> M = tM + r + 8*half ; N = tN + (lane & 15)
    int n = tN + lr;
#pragma unroll
    for (int r = 0; r < 8; ++r) {
        int m = tM + r + 8 * half;
        C[(size_t)m * Nn + n] = acc[r];
    }
}

// ---------------- per-node attention scores ----------------
__global__ void k_att_scores(const float* __restrict__ h,
                             const float* __restrict__ att_src,
                             const float* __restrict__ att_dst,
                             float* __restrict__ a_s,
                             float* __restrict__ a_d) {
    int i = blockIdx.x * blockDim.x + threadIdx.x;     // (node, head)
    if (i >= NN * 2) return;
    int node = i >> 1, hd = i & 1;
    const float* hp = h + (size_t)node * HC + hd * CC;
    const float* as = att_src + hd * CC;
    const float* ad = att_dst + hd * CC;
    float s = 0.f, d = 0.f;
#pragma unroll 4
    for (int c = 0; c < CC; ++c) { float v = hp[c]; s += v * as[c]; d += v * ad[c]; }
    a_s[i] = s; a_d[i] = d;
}

// ---------------- edge phase ----------------
__global__ void k_edge_max(const int* __restrict__ ei,
                           const float* __restrict__ a_s,
                           const float* __restrict__ a_d,
                           unsigned* __restrict__ m_enc) {
    int e = blockIdx.x * blockDim.x + threadIdx.x;
    if (e >= EP) return;
    int s, d; edge_sd(e, ei, s, d);
#pragma unroll
    for (int h = 0; h < 2; ++h) {
        float l = a_s[s * 2 + h] + a_d[d * 2 + h];
        l = (l > 0.f) ? l : NEG_SLOPE * l;             // leaky_relu
        atomicMax(&m_enc[d * 2 + h], enc_f32(l));
    }
}

__global__ void k_edge_exp(const int* __restrict__ ei,
                           const float* __restrict__ a_s,
                           const float* __restrict__ a_d,
                           const unsigned* __restrict__ m_enc,
                           float* __restrict__ e_buf,
                           float* __restrict__ denom) {
    int e = blockIdx.x * blockDim.x + threadIdx.x;
    if (e >= EP) return;
    int s, d; edge_sd(e, ei, s, d);
#pragma unroll
    for (int h = 0; h < 2; ++h) {
        float l = a_s[s * 2 + h] + a_d[d * 2 + h];
        l = (l > 0.f) ? l : NEG_SLOPE * l;
        float m  = dec_f32(m_enc[d * 2 + h]);
        float ex = expf(l - m);
        e_buf[(size_t)e * 2 + h] = ex;
        atomicAdd(&denom[d * 2 + h], ex);
    }
}

// block per edge, one thread per output channel (256 == HC) -> coalesced
__global__ __launch_bounds__(HC)
void k_edge_scatter(const int* __restrict__ ei,
                    const float* __restrict__ h,
                    const float* __restrict__ e_buf,
                    const float* __restrict__ denom,
                    float* __restrict__ acc) {
    int e = blockIdx.x;
    int c = threadIdx.x;               // 0..255
    int s, d; edge_sd(e, ei, s, d);
    int hd = c >> 7;                   // head = channel / 128
    float alpha = e_buf[(size_t)e * 2 + hd] / (denom[d * 2 + hd] + 1e-16f);
    atomicAdd(&acc[(size_t)d * HC + c], alpha * h[(size_t)s * HC + c]);
}

// ---------------- between layers: h = tanh(acc + b); also make bf16 copy ----
__global__ void k_post_tanh(const float* __restrict__ acc,
                            const float* __restrict__ b,
                            float* __restrict__ hout,
                            __bf16* __restrict__ hbf, int total) {
    int i = blockIdx.x * blockDim.x + threadIdx.x;
    if (i < total) {
        float v = tanhf(acc[i] + b[i & (HC - 1)]);
        hout[i] = v;
        hbf[i]  = (__bf16)v;
    }
}

// ---------------- final linear + softmax ----------------
__global__ void k_final(const float* __restrict__ acc,
                        const float* __restrict__ b1,
                        const float* __restrict__ Wl,
                        const float* __restrict__ bl,
                        float* __restrict__ out) {
    int node = blockIdx.x * blockDim.x + threadIdx.x;
    if (node >= NN) return;
    const float* a = acc + (size_t)node * HC;
    float z0 = bl[0], z1 = bl[1];
#pragma unroll 4
    for (int k = 0; k < HC; ++k) {
        float v = a[k] + b1[k];
        z0 += v * Wl[k * 2 + 0];
        z1 += v * Wl[k * 2 + 1];
    }
    float mx = fmaxf(z0, z1);
    float e0 = expf(z0 - mx), e1 = expf(z1 - mx);
    float inv = 1.f / (e0 + e1);
    out[node * 2 + 0] = e0 * inv;
    out[node * 2 + 1] = e1 * inv;
}

// ---------------- host launch ----------------
extern "C" void kernel_launch(void* const* d_in, const int* in_sizes, int n_in,
                              void* d_out, int out_size, void* d_ws, size_t ws_size,
                              hipStream_t stream) {
    const float* x        = (const float*)d_in[0];
    const int*   ei       = (const int*)  d_in[1];
    const float* W0       = (const float*)d_in[2];
    const float* att_src0 = (const float*)d_in[3];
    const float* att_dst0 = (const float*)d_in[4];
    const float* b0       = (const float*)d_in[5];
    const float* W1       = (const float*)d_in[6];
    const float* att_src1 = (const float*)d_in[7];
    const float* att_dst1 = (const float*)d_in[8];
    const float* b1       = (const float*)d_in[9];
    const float* Wl       = (const float*)d_in[10];
    const float* bl       = (const float*)d_in[11];
    float* out = (float*)d_out;

    // ---- workspace carve-up (256B aligned) ----
    char* base = (char*)d_ws;
    size_t off = 0;
    auto carve = [&](size_t bytes) -> char* {
        char* p = base + off;
        off = (off + bytes + 255) & ~(size_t)255;
        return p;
    };
    float*    h_feat = (float*)   carve((size_t)NN * HC * 4);   // features / gemm out
    float*    accb   = (float*)   carve((size_t)NN * HC * 4);   // segment-sum accumulator
    __bf16*   xbf    = (__bf16*)  carve((size_t)NN * HC * 2);   // bf16 activations
    __bf16*   wtbf   = (__bf16*)  carve((size_t)HC * HC * 2);   // bf16 transposed weights
    float*    a_s    = (float*)   carve((size_t)NN * 2 * 4);
    float*    a_d    = (float*)   carve((size_t)NN * 2 * 4);
    unsigned* m_enc  = (unsigned*)carve((size_t)NN * 2 * 4);
    float*    denomb = (float*)   carve((size_t)NN * 2 * 4);
    float*    e_buf  = (float*)   carve((size_t)EP * 2 * 4);
    (void)ws_size; (void)in_sizes; (void)n_in; (void)out_size;

    const int T = 256;
    dim3 blk(T);

    // =================== layer 0 ===================
    k_f32_to_bf16<<<(NN * DD + T - 1) / T, blk, 0, stream>>>(x, xbf, NN * DD);
    k_transpose_bf16<<<(DD * HC + T - 1) / T, blk, 0, stream>>>(W0, wtbf, DD, HC);

    // waves = (NN/16)*(HC/16) = 3125*16 = 50000 ; 8 waves/block
    k_gemm_bf16_wmma<<<(NN / 16) * (HC / 16) / 8, blk, 0, stream>>>(
        xbf, wtbf, h_feat, NN, HC, DD);

    k_att_scores<<<(NN * 2 + T - 1) / T, blk, 0, stream>>>(
        h_feat, att_src0, att_dst0, a_s, a_d);

    hipMemsetAsync(accb,   0, (size_t)NN * HC * 4, stream);
    hipMemsetAsync(m_enc,  0, (size_t)NN * 2 * 4, stream);   // 0 == encoded minimum
    hipMemsetAsync(denomb, 0, (size_t)NN * 2 * 4, stream);

    k_edge_max    <<<(EP + T - 1) / T, blk, 0, stream>>>(ei, a_s, a_d, m_enc);
    k_edge_exp    <<<(EP + T - 1) / T, blk, 0, stream>>>(ei, a_s, a_d, m_enc, e_buf, denomb);
    k_edge_scatter<<<EP, dim3(HC), 0, stream>>>(ei, h_feat, e_buf, denomb, accb);

    k_post_tanh<<<(NN * HC + T - 1) / T, blk, 0, stream>>>(accb, b0, h_feat, xbf, NN * HC);

    // =================== layer 1 ===================
    k_transpose_bf16<<<(HC * HC + T - 1) / T, blk, 0, stream>>>(W1, wtbf, HC, HC);

    k_gemm_bf16_wmma<<<(NN / 16) * (HC / 16) / 8, blk, 0, stream>>>(
        xbf, wtbf, h_feat, NN, HC, HC);

    k_att_scores<<<(NN * 2 + T - 1) / T, blk, 0, stream>>>(
        h_feat, att_src1, att_dst1, a_s, a_d);

    hipMemsetAsync(accb,   0, (size_t)NN * HC * 4, stream);
    hipMemsetAsync(m_enc,  0, (size_t)NN * 2 * 4, stream);
    hipMemsetAsync(denomb, 0, (size_t)NN * 2 * 4, stream);

    k_edge_max    <<<(EP + T - 1) / T, blk, 0, stream>>>(ei, a_s, a_d, m_enc);
    k_edge_exp    <<<(EP + T - 1) / T, blk, 0, stream>>>(ei, a_s, a_d, m_enc, e_buf, denomb);
    k_edge_scatter<<<EP, dim3(HC), 0, stream>>>(ei, h_feat, e_buf, denomb, accb);

    // =================== classifier + softmax ===================
    k_final<<<(NN + T - 1) / T, blk, 0, stream>>>(accb, b1, Wl, bl, out);
}
